// SimpleLSTM_58609123721295
// MI455X (gfx1250) — compile-verified, matrix operand order checked by reading, not run on hardware
//
#include <hip/hip_runtime.h>
#include <stdint.h>

// ---------------- types ----------------
typedef __attribute__((ext_vector_type(16))) __bf16        v16bf;
typedef __attribute__((ext_vector_type(8)))  float         v8f;
typedef __attribute__((ext_vector_type(4)))  unsigned int  u32x4;

// ---------------- problem dims ----------------
// I=512, H=1024, B=64, S=512, G=4H=4096, M=B*S=32768
constexpr int    KI   = 512;
constexpr int    KH   = 1024;
constexpr int    NB_B = 64;
constexpr int    SEQ  = 512;
constexpr int    GATE = 4096;
constexpr size_t MROW = 32768;   // B*S

// LDS slab for the scan: 128 Wh^T rows, padded row stride to avoid bank conflicts
constexpr int    LDSP      = KH + 8;                 // 1032 bf16 per row (2064 B)
constexpr int    SCAN_BLKS = 32;                     // 4096 cols / 128 per block
constexpr int    SMEM_B    = 128 * LDSP * 2;         // 264,192 bytes (<320KB WGP LDS)

// ---------------- workspace layout (bytes) ----------------
constexpr size_t OFF_XG  = 0;                       // xg bf16 [32768][4096]
constexpr size_t SZ_XG   = MROW * (size_t)GATE * 2;
constexpr size_t OFF_XB  = OFF_XG + SZ_XG;          // x bf16 [32768][512]
constexpr size_t SZ_XB   = MROW * (size_t)KI * 2;
constexpr size_t OFF_WXT = OFF_XB + SZ_XB;          // Wx^T bf16 [4096][512]
constexpr size_t SZ_WXT  = (size_t)GATE * KI * 2;
constexpr size_t OFF_WHT = OFF_WXT + SZ_WXT;        // Wh^T bf16 [4096][1024]
constexpr size_t SZ_WHT  = (size_t)GATE * KH * 2;
constexpr size_t OFF_HB  = OFF_WHT + SZ_WHT;        // h bf16 [64][1024]
constexpr size_t SZ_HB   = (size_t)NB_B * KH * 2;
constexpr size_t OFF_C   = OFF_HB + SZ_HB;          // c fp32 [64][1024]
constexpr size_t SZ_C    = (size_t)NB_B * KH * 4;
constexpr size_t OFF_GT  = OFF_C + SZ_C;            // gates fp32 [64][4096]
constexpr size_t SZ_GT   = (size_t)NB_B * GATE * 4;
constexpr size_t OFF_BAR = OFF_GT + SZ_GT;          // barrier ints

constexpr size_t OUT_HC  = (size_t)NB_B * SEQ * KH; // 33554432: start of (h,c) tail

// ---------------- bf16 helpers (explicit RNE) ----------------
__device__ __forceinline__ __bf16 f2bf(float f) {
  union { float f; uint32_t u; } a; a.f = f;
  uint32_t r = a.u + 0x7FFFu + ((a.u >> 16) & 1u);
  union { uint16_t u; __bf16 b; } o; o.u = (uint16_t)(r >> 16);
  return o.b;
}
__device__ __forceinline__ float bf2f(__bf16 b) {
  union { uint16_t u; __bf16 b; } i; i.b = b;
  union { uint32_t u; float f; } o; o.u = ((uint32_t)i.u) << 16;
  return o.f;
}

__device__ __forceinline__ float sigmoidf(float x) {
  return 1.0f / (1.0f + __expf(-x));
}
__device__ __forceinline__ float tanh_fast(float x) {
  // tanh(x) = 1 - 2/(exp(2x)+1)
  return 1.0f - 2.0f / (__expf(2.0f * x) + 1.0f);
}

// ---------------- WMMA fragment loads ----------------
// 16-bit A-matrix 16x32 layout (ISA 7.12.2): lanes 0-15 hold row M=lane with
// K = {k0..k0+7} in v0..v3 and {k0+16..k0+23} in v4..v7; lanes 16-31 hold the
// same rows with K offset +8 / +24. B uses the symmetric layout with the
// matrix stored N-major (we pre-transpose W), so both are two b128 loads.
__device__ __forceinline__ v16bf load_frag(const __bf16* __restrict__ base,
                                           int ld, int row0, int k0, int lane) {
  const int r = row0 + (lane & 15);
  const int k = k0 + ((lane >> 4) << 3);
  const __bf16* p = base + (size_t)r * ld + k;
  union { v16bf v; struct { u32x4 lo, hi; } q; } f;
  f.q.lo = *reinterpret_cast<const u32x4*>(p);
  f.q.hi = *reinterpret_cast<const u32x4*>(p + 16);
  return f.v;
}

// Same fragment shape sourced from the padded LDS slab (ds_load_b128 x2).
__device__ __forceinline__ v16bf load_frag_lds(const __bf16* sm,
                                               int row0, int k0, int lane) {
  const int r = row0 + (lane & 15);
  const int k = k0 + ((lane >> 4) << 3);
  const __bf16* p = sm + r * LDSP + k;
  union { v16bf v; struct { u32x4 lo, hi; } q; } f;
  f.q.lo = *reinterpret_cast<const u32x4*>(p);
  f.q.hi = *reinterpret_cast<const u32x4*>(p + 16);
  return f.v;
}

// ---------------- grid-wide barrier (persistent scan kernel) ----------------
__device__ __forceinline__ void grid_barrier(int* counter, int* sense,
                                             int nblk, int& local) {
  __threadfence();
  __syncthreads();
  if (threadIdx.x == 0) {
    int s = local ^ 1;
    local = s;
    if (atomicAdd(counter, 1) == nblk - 1) {
      *counter = 0;
      __threadfence();
      __hip_atomic_store(sense, s, __ATOMIC_RELEASE, __HIP_MEMORY_SCOPE_AGENT);
    } else {
      while (__hip_atomic_load(sense, __ATOMIC_ACQUIRE,
                               __HIP_MEMORY_SCOPE_AGENT) != s) {
        __builtin_amdgcn_s_sleep(1);
      }
    }
  }
  __syncthreads();
}

// ---------------- prep kernels ----------------
__global__ void k_cvt_x(const float* __restrict__ x, __bf16* __restrict__ xb, int n) {
  int i = blockIdx.x * blockDim.x + threadIdx.x;
  if (i < n) xb[i] = f2bf(x[i]);
}

// W is (1536 x 4096) row-major fp32. Split into Wx^T (4096x512) and Wh^T (4096x1024), bf16.
__global__ void k_prep_w(const float* __restrict__ W,
                         __bf16* __restrict__ wxt, __bf16* __restrict__ wht) {
  int i = blockIdx.x * blockDim.x + threadIdx.x;
  if (i >= 1536 * 4096) return;
  int k = i >> 12;        // row in W
  int n = i & 4095;       // col in W
  __bf16 v = f2bf(W[i]);
  if (k < KI) wxt[(size_t)n * KI + k] = v;
  else        wht[(size_t)n * KH + (k - KI)] = v;
}

__global__ void k_init(__bf16* __restrict__ hb, float* __restrict__ c, int* bar) {
  int i = blockIdx.x * blockDim.x + threadIdx.x;
  if (i < NB_B * KH) { hb[i] = f2bf(0.0f); c[i] = 0.0f; }
  if (i < 2) bar[i] = 0;
}

// ---------------- phase 1: xg = x @ Wx + b  (bf16 WMMA, fp32 accum) ----------------
// Each wave computes a 64x64 tile: 4x4 grid of 16x16 WMMA accumulators.
__global__ void __launch_bounds__(256)
k_gemm_xg(const __bf16* __restrict__ xb,   // [32768][512]
          const __bf16* __restrict__ wxt,  // [4096][512] (N-major)
          const float*  __restrict__ bias, // [4096]
          __bf16* __restrict__ xg)         // [32768][4096]
{
  const int lane = threadIdx.x & 31;
  const int wave = blockIdx.x * (blockDim.x >> 5) + (threadIdx.x >> 5);
  const int m0 = (wave >> 6) * 64;   // 512 m-blocks
  const int n0 = (wave & 63) * 64;   // 64  n-blocks

  v8f acc[4][4];
#pragma unroll
  for (int i = 0; i < 4; ++i)
#pragma unroll
    for (int j = 0; j < 4; ++j) { v8f z{}; acc[i][j] = z; }

  for (int k = 0; k < KI; k += 32) {
    v16bf A[4], Bf[4];
#pragma unroll
    for (int i = 0; i < 4; ++i) A[i]  = load_frag(xb,  KI, m0 + 16 * i, k, lane);
#pragma unroll
    for (int j = 0; j < 4; ++j) Bf[j] = load_frag(wxt, KI, n0 + 16 * j, k, lane);
#pragma unroll
    for (int i = 0; i < 4; ++i)
#pragma unroll
      for (int j = 0; j < 4; ++j)
        acc[i][j] = __builtin_amdgcn_wmma_f32_16x16x32_bf16(
            false, A[i], false, Bf[j], (short)0, acc[i][j], false, false);
  }

  const int mrow = (lane >> 4) << 3;  // +8 for upper half-lanes
  const int ncol = lane & 15;
#pragma unroll
  for (int j = 0; j < 4; ++j) {
    const int n = n0 + 16 * j + ncol;
    const float bj = bias[n];
#pragma unroll
    for (int i = 0; i < 4; ++i) {
#pragma unroll
      for (int r = 0; r < 8; ++r) {
        const int m = m0 + 16 * i + r + mrow;
        xg[(size_t)m * GATE + n] = f2bf(acc[i][j][r] + bj);
      }
    }
  }
}

// ---------------- phase 2: persistent LSTM scan ----------------
// 32 blocks x 256 threads = 256 waves. Block b owns gate columns
// [128b, 128b+128): its Wh^T slab (128 x 1024 bf16, padded) is staged into
// LDS ONCE, then reused by all 512 time steps (CDNA5: 320KB LDS per WGP).
// Wave w in block: m-tile = (w&3)*16, column half = (w>>2)*64 -> 16x64 tile.
// Per step: gates = h @ Wh + xg[:,t,:], grid barrier, elementwise LSTM cell
// update (sigmoid/tanh, c/h update, write outputs), grid barrier.
__global__ void __launch_bounds__(256)
k_scan(const __bf16* __restrict__ xg,   // [32768][4096] (row = b*512 + t)
       const __bf16* __restrict__ wht,  // [4096][1024] (N-major)
       __bf16* __restrict__ hb,         // [64][1024]
       float*  __restrict__ c,          // [64][1024]
       float*  __restrict__ gates,      // [64][4096]
       float*  __restrict__ out,        // outputs | h | c
       int* bar)
{
  extern __shared__ __bf16 smem[];                // [128][LDSP]

  const int tid   = threadIdx.x;
  const int lane  = tid & 31;
  const int wv    = tid >> 5;                     // 0..7
  const int mt    = (wv & 3) << 4;                // 0,16,32,48
  const int nloc  = (wv >> 2) << 6;               // 0 or 64 (local column base)
  const int nglb  = blockIdx.x * 128 + nloc;      // global column base
  const int gtid  = blockIdx.x * 256 + tid;       // 0..8191
  const int nthr  = gridDim.x * 256;
  int sense_local = 0;

  // ---- stage this block's Wh^T slab into LDS (once) ----
  {
    const __bf16* src = wht + (size_t)blockIdx.x * 128 * KH;
    for (int idx = tid; idx < 128 * (KH / 8); idx += blockDim.x) {
      const int row = idx >> 7;            // /(KH/8)=128 chunks per row
      const int kc  = (idx & 127) << 3;    // 8 bf16 = 16B chunks
      *reinterpret_cast<u32x4*>(&smem[row * LDSP + kc]) =
          *reinterpret_cast<const u32x4*>(&src[(size_t)row * KH + kc]);
    }
  }
  __syncthreads();

  for (int t = 0; t < SEQ; ++t) {
    // ---- gates tile: h @ Wh (A from global/L2, B from LDS) ----
    v8f acc[4];
#pragma unroll
    for (int j = 0; j < 4; ++j) { v8f z{}; acc[j] = z; }

    for (int k = 0; k < KH; k += 32) {
      v16bf a = load_frag(hb, KH, mt, k, lane);
#pragma unroll
      for (int j = 0; j < 4; ++j) {
        v16bf bfr = load_frag_lds(smem, nloc + 16 * j, k, lane);
        acc[j] = __builtin_amdgcn_wmma_f32_16x16x32_bf16(
            false, a, false, bfr, (short)0, acc[j], false, false);
      }
    }

    const int mrow = (lane >> 4) << 3;
    const int ncol = lane & 15;
#pragma unroll
    for (int j = 0; j < 4; ++j) {
      const int n = nglb + 16 * j + ncol;
#pragma unroll
      for (int r = 0; r < 8; ++r) {
        const int m = mt + r + mrow;
        gates[m * GATE + n] =
            acc[j][r] + bf2f(xg[((size_t)m * SEQ + t) * GATE + n]);
      }
    }

    grid_barrier(bar, bar + 1, gridDim.x, sense_local);

    // ---- elementwise cell update ----
    for (int e = gtid; e < NB_B * KH; e += nthr) {
      const int b = e >> 10;
      const int j = e & 1023;
      const float* gr = gates + b * GATE;
      const float fg = sigmoidf(gr[j]);
      const float ig = sigmoidf(gr[KH + j]);
      const float gg = tanh_fast(gr[2 * KH + j]);
      const float og = sigmoidf(gr[3 * KH + j]);
      const float cn = fg * c[e] + ig * gg;
      c[e] = cn;
      const float hn = og * tanh_fast(cn);
      out[((size_t)b * SEQ + t) * KH + j] = hn;
      hb[e] = f2bf(hn);
      if (t == SEQ - 1) {
        out[OUT_HC + e] = hn;                     // final h
        out[OUT_HC + (size_t)NB_B * KH + e] = cn; // final c
      }
    }

    grid_barrier(bar, bar + 1, gridDim.x, sense_local);
  }
}

// ---------------- host launcher ----------------
extern "C" void kernel_launch(void* const* d_in, const int* in_sizes, int n_in,
                              void* d_out, int out_size, void* d_ws, size_t ws_size,
                              hipStream_t stream) {
  const float* x = (const float*)d_in[0];   // (64, 512, 512)
  const float* W = (const float*)d_in[1];   // (1536, 4096)
  const float* b = (const float*)d_in[2];   // (4096,)
  float* out = (float*)d_out;
  char*  ws  = (char*)d_ws;

  __bf16* xg  = (__bf16*)(ws + OFF_XG);
  __bf16* xb  = (__bf16*)(ws + OFF_XB);
  __bf16* wxt = (__bf16*)(ws + OFF_WXT);
  __bf16* wht = (__bf16*)(ws + OFF_WHT);
  __bf16* hb  = (__bf16*)(ws + OFF_HB);
  float*  c   = (float*)(ws + OFF_C);
  float*  gt  = (float*)(ws + OFF_GT);
  int*    bar = (int*)(ws + OFF_BAR);

  const int nx = (int)(MROW * KI);          // 16,777,216
  const int nw = 1536 * 4096;               //  6,291,456

  // Allow >64KB dynamic LDS for the scan kernel (CDNA5 WGP has 320KB).
  static bool attr_set = [] {
    (void)hipFuncSetAttribute((const void*)k_scan,
                              hipFuncAttributeMaxDynamicSharedMemorySize, SMEM_B);
    return true;
  }();
  (void)attr_set;

  k_cvt_x<<<(nx + 255) / 256, 256, 0, stream>>>(x, xb, nx);
  k_prep_w<<<(nw + 255) / 256, 256, 0, stream>>>(W, wxt, wht);
  k_init<<<(NB_B * KH + 255) / 256, 256, 0, stream>>>(hb, c, bar);

  // 512 m-blocks * 64 n-blocks = 32768 waves = 4096 blocks of 8 waves
  k_gemm_xg<<<4096, 256, 0, stream>>>(xb, wxt, b, xg);

  // persistent scan: 32 co-resident blocks, 8 waves each, 264KB LDS slab
  k_scan<<<SCAN_BLKS, 256, SMEM_B, stream>>>(xg, wht, hb, c, gt, out, bar);
}
